// SelfAttention_57535381897492
// MI455X (gfx1250) — compile-verified, matrix operand order checked by reading, not run on hardware
//
#include <hip/hip_runtime.h>

// Problem constants (fixed by the reference)
#define SEQ    2048
#define BATCH  2
#define HEADS  16
#define HDIM   64
#define DMODEL 1024
#define ROWS   (BATCH * SEQ)     // 4096
#define N_QKV  (3 * DMODEL)      // 3072

typedef __attribute__((ext_vector_type(16))) __bf16        v16bf;
typedef __attribute__((ext_vector_type(8)))  float         v8f;
typedef __attribute__((ext_vector_type(4)))  unsigned int  u32x4;

__device__ __forceinline__ unsigned short f2bf(float f) {
  unsigned int u = __float_as_uint(f);
  u += 0x7FFFu + ((u >> 16) & 1u);            // round-to-nearest-even
  return (unsigned short)(u >> 16);
}
__device__ __forceinline__ float bf2f(unsigned short h) {
  return __uint_as_float(((unsigned int)h) << 16);
}

// Load one 16x32 bf16 WMMA fragment (A-layout; B uses the same layout when the
// source matrix is stored [N x K] row-major).  p = start of this lane's row.
// Per ISA 7.12.2: half 0 holds K = 0..7 and 16..23, half 1 holds K = 8..15, 24..31.
__device__ __forceinline__ v16bf frag_ld(const unsigned short* p, int half) {
  union { u32x4 q[2]; v16bf v; } u;
  u.q[0] = *(const u32x4*)(p + half * 8);
  u.q[1] = *(const u32x4*)(p + 16 + half * 8);
  return u.v;
}

__device__ __forceinline__ v8f wmma_bf16(v16bf a, v16bf b, v8f c) {
  // (neg_a, A, neg_b, B, c_mod, C, reuse_a, reuse_b)
  return __builtin_amdgcn_wmma_f32_16x16x32_bf16(false, a, false, b, (short)0, c,
                                                 false, false);
}

// ---- CDNA5 async memory->LDS copy (ASYNCcnt-tracked, no VGPR data) ---------
__device__ __forceinline__ void async_g2l_b128(const unsigned short* gsrc,
                                               unsigned short* ldst) {
  unsigned int loff = (unsigned int)(unsigned long long)(uintptr_t)ldst;
  asm volatile("global_load_async_to_lds_b128 %0, %1, off"
               :: "v"(loff), "v"(gsrc) : "memory");
}
__device__ __forceinline__ void wait_async0() {
#if __has_builtin(__builtin_amdgcn_s_wait_asynccnt)
  __builtin_amdgcn_s_wait_asynccnt(0);
#else
  asm volatile("s_wait_asynccnt 0x0" ::: "memory");
#endif
}

// ---------------------------------------------------------------------------
// Kernel 1: fp32 -> bf16 convert of x; convert + transpose Wqkv, Wo to [N x K]
// ---------------------------------------------------------------------------
__global__ void prep_kernel(const float* __restrict__ x,
                            const float* __restrict__ wqkv,
                            const float* __restrict__ wo,
                            unsigned short* __restrict__ xb,
                            unsigned short* __restrict__ wqkvT,
                            unsigned short* __restrict__ woT) {
  const int NX  = ROWS * DMODEL;     // 4194304
  const int NW1 = N_QKV * DMODEL;    // 3145728
  const int NW2 = DMODEL * DMODEL;   // 1048576
  int idx = blockIdx.x * blockDim.x + threadIdx.x;
  if (idx >= NX + NW1 + NW2) return;
  if (idx < NX) {
    xb[idx] = f2bf(x[idx]);
  } else if (idx < NX + NW1) {
    int t = idx - NX;
    int n = t >> 10, k = t & 1023;                 // wqkvT[n][k] = Wqkv[k][n]
    wqkvT[t] = f2bf(wqkv[(size_t)k * N_QKV + n]);
  } else {
    int t = idx - NX - NW1;
    int n = t >> 10, k = t & 1023;                 // woT[n][k] = Wo[k][n]
    woT[t] = f2bf(wo[(size_t)k * DMODEL + n]);
  }
}

// ---------------------------------------------------------------------------
// Tiled bf16 GEMM: C[ROWS x N] = A[ROWS x 1024] * Bt[N x 1024]^T
//   128x128 block, 8 waves (4x2), wave tile 32x64, K step 32.
//   Double-buffered LDS tiles filled by global_load_async_to_lds_b128.
// MODE 0: scatter into q/k [B,H,S,Hd] bf16 and vT [B,H,Hd,S] bf16
// MODE 1: plain f32 row-major store
// ---------------------------------------------------------------------------
template <int MODE>
__global__ __launch_bounds__(256) void gemm_kernel(
    const unsigned short* __restrict__ A,
    const unsigned short* __restrict__ Bt,
    int N,
    unsigned short* __restrict__ qO,
    unsigned short* __restrict__ kO,
    unsigned short* __restrict__ vTO,
    float* __restrict__ out) {
  __shared__ __align__(16) unsigned short sA[2][128 * 40];
  __shared__ __align__(16) unsigned short sB[2][128 * 40];

  const int tid  = threadIdx.x;
  const int wave = tid >> 5, lane = tid & 31;
  const int wm = wave & 3, wn = wave >> 2;
  const int r = lane & 15, half = lane >> 4;
  const int m0 = blockIdx.y * 128, n0 = blockIdx.x * 128;

  // each thread copies 2 slots of 8 bf16 per tile per matrix
  const int row0 = tid >> 2, kc0 = (tid & 3) << 3;        // slot = tid
  const int row1 = (tid + 256) >> 2, kc1 = kc0;           // slot = tid + 256

  const v8f zf = {0.f, 0.f, 0.f, 0.f, 0.f, 0.f, 0.f, 0.f};
  v8f acc[2][4];
#pragma unroll
  for (int f = 0; f < 2; ++f)
#pragma unroll
    for (int g = 0; g < 4; ++g) acc[f][g] = zf;

  const int NK = DMODEL / 32;  // 32 K-chunks

  // prologue: stage chunk 0 into buffer 0
  {
    const int k0 = 0;
    async_g2l_b128(&A [(size_t)(m0 + row0) * DMODEL + k0 + kc0], &sA[0][row0 * 40 + kc0]);
    async_g2l_b128(&A [(size_t)(m0 + row1) * DMODEL + k0 + kc1], &sA[0][row1 * 40 + kc1]);
    async_g2l_b128(&Bt[(size_t)(n0 + row0) * DMODEL + k0 + kc0], &sB[0][row0 * 40 + kc0]);
    async_g2l_b128(&Bt[(size_t)(n0 + row1) * DMODEL + k0 + kc1], &sB[0][row1 * 40 + kc1]);
  }

  for (int kt = 0; kt < NK; ++kt) {
    const int cur = kt & 1, nxt = cur ^ 1;
    wait_async0();       // my async copies for buffer `cur` (and earlier) landed
    __syncthreads();     // everyone's copies landed

    if (kt + 1 < NK) {   // kick off next chunk into the other buffer
      const int k0 = (kt + 1) * 32;
      async_g2l_b128(&A [(size_t)(m0 + row0) * DMODEL + k0 + kc0], &sA[nxt][row0 * 40 + kc0]);
      async_g2l_b128(&A [(size_t)(m0 + row1) * DMODEL + k0 + kc1], &sA[nxt][row1 * 40 + kc1]);
      async_g2l_b128(&Bt[(size_t)(n0 + row0) * DMODEL + k0 + kc0], &sB[nxt][row0 * 40 + kc0]);
      async_g2l_b128(&Bt[(size_t)(n0 + row1) * DMODEL + k0 + kc1], &sB[nxt][row1 * 40 + kc1]);
    }

    v16bf aF[2], bF[4];
#pragma unroll
    for (int f = 0; f < 2; ++f)
      aF[f] = frag_ld(&sA[cur][(wm * 32 + f * 16 + r) * 40], half);
#pragma unroll
    for (int g = 0; g < 4; ++g)
      bF[g] = frag_ld(&sB[cur][(wn * 64 + g * 16 + r) * 40], half);
#pragma unroll
    for (int f = 0; f < 2; ++f)
#pragma unroll
      for (int g = 0; g < 4; ++g) acc[f][g] = wmma_bf16(aF[f], bF[g], acc[f][g]);

    __syncthreads();     // all reads of `cur` done before it is refilled
  }

  // C-fragment layout: col = lane%16, row = v + 8*(lane/16)
#pragma unroll
  for (int f = 0; f < 2; ++f)
#pragma unroll
    for (int g = 0; g < 4; ++g)
#pragma unroll
      for (int v = 0; v < 8; ++v) {
        int m = m0 + wm * 32 + f * 16 + v + 8 * half;
        int c = n0 + wn * 64 + g * 16 + r;
        float val = acc[f][g][v];
        if (MODE == 1) {
          out[(size_t)m * N + c] = val;
        } else {
          int sel = c >> 10, cc = c & 1023;
          int h = cc >> 6, d = cc & 63;
          int b = m >> 11, sIdx = m & 2047;
          unsigned short bv = f2bf(val);
          if (sel == 0)
            qO[((size_t)(b * HEADS + h) * SEQ + sIdx) * HDIM + d] = bv;
          else if (sel == 1)
            kO[((size_t)(b * HEADS + h) * SEQ + sIdx) * HDIM + d] = bv;
          else  // V stored transposed: [B,H,Hd,S]
            vTO[((size_t)(b * HEADS + h) * HDIM + d) * SEQ + sIdx] = bv;
        }
      }
}

// ---------------------------------------------------------------------------
// Kernel 3: RoPE in place on q,k bf16 [B,H,S,Hd].  One thread per (row, d<32).
// ---------------------------------------------------------------------------
__global__ void rope_kernel(unsigned short* __restrict__ q,
                            unsigned short* __restrict__ k) {
  int idx = blockIdx.x * blockDim.x + threadIdx.x;  // BATCH*HEADS*SEQ*32 total
  int d = idx & 31;
  int row = idx >> 5;
  int s = row & (SEQ - 1);
  // inv_freq = 10000^(-d/32) = exp(-d * ln(10000)/32)
  float freq = __expf(-(float)d * 0.28782300114517747f);
  float ang = (float)s * freq;
  float sn, cs;
  __sincosf(ang, &sn, &cs);
  size_t base = (size_t)row * HDIM + d;
  {
    float x1 = bf2f(q[base]), x2 = bf2f(q[base + 32]);
    q[base]      = f2bf(x1 * cs - x2 * sn);
    q[base + 32] = f2bf(x2 * cs + x1 * sn);
  }
  {
    float x1 = bf2f(k[base]), x2 = bf2f(k[base + 32]);
    k[base]      = f2bf(x1 * cs - x2 * sn);
    k[base + 32] = f2bf(x2 * cs + x1 * sn);
  }
}

// ---------------------------------------------------------------------------
// Kernel 4: causal flash attention.  Block = 64 q-rows (4 waves x 16 rows),
// grid (S/64, B*H).  Scores and P@V both via v_wmma_f32_16x16x32_bf16.
// ---------------------------------------------------------------------------
__global__ __launch_bounds__(128) void flash_kernel(
    const unsigned short* __restrict__ q,
    const unsigned short* __restrict__ k,
    const unsigned short* __restrict__ vT,
    unsigned short* __restrict__ attnOut) {
  __shared__ __align__(16) unsigned short sP[4 * 16 * 40];  // per-wave P tiles

  const int tid  = threadIdx.x;
  const int wave = tid >> 5, lane = tid & 31;
  const int r = lane & 15, half = lane >> 4;
  const int q0 = blockIdx.x * 64;
  const int bh = blockIdx.y;

  const unsigned short* qp = q  + (size_t)bh * SEQ * HDIM;
  const unsigned short* kp = k  + (size_t)bh * SEQ * HDIM;
  const unsigned short* vp = vT + (size_t)bh * HDIM * SEQ;

  // Q fragments for this wave's 16 rows (Hd = 64 -> two K=32 fragments)
  const int qr = q0 + wave * 16 + r;
  v16bf qf[2];
#pragma unroll
  for (int f = 0; f < 2; ++f)
    qf[f] = frag_ld(qp + (size_t)qr * HDIM + f * 32, half);

  const v8f zf = {0.f, 0.f, 0.f, 0.f, 0.f, 0.f, 0.f, 0.f};
  v8f o[4];
#pragma unroll
  for (int g = 0; g < 4; ++g) o[g] = zf;
  float mrow[8], lrow[8];
#pragma unroll
  for (int v = 0; v < 8; ++v) { mrow[v] = -3.0e38f; lrow[v] = 0.0f; }

  unsigned short* pw = &sP[wave * 16 * 40];
  const int nChunks = (q0 + 64) >> 5;  // uniform across all 4 waves

  for (int jc = 0; jc < nChunks; ++jc) {
    const int j0 = jc << 5;
    // --- scores: S[16q x 32j] = Q(16x64) . K^T ---
    v8f s0 = zf, s1 = zf;
#pragma unroll
    for (int db = 0; db < 2; ++db) {
      v16bf kb0 = frag_ld(kp + (size_t)(j0 + r) * HDIM + db * 32, half);
      v16bf kb1 = frag_ld(kp + (size_t)(j0 + 16 + r) * HDIM + db * 32, half);
      s0 = wmma_bf16(qf[db], kb0, s0);
      s1 = wmma_bf16(qf[db], kb1, s1);
    }
    // --- online softmax (scale 1/sqrt(64), causal mask, 16-lane reductions) ---
#pragma unroll
    for (int v = 0; v < 8; ++v) {
      int i = q0 + wave * 16 + v + 8 * half;
      float e0 = s0[v] * 0.125f;
      float e1 = s1[v] * 0.125f;
      if (j0 + r > i)      e0 = -3.0e38f;
      if (j0 + 16 + r > i) e1 = -3.0e38f;
      float mx = fmaxf(e0, e1);
#pragma unroll
      for (int off = 1; off < 16; off <<= 1)
        mx = fmaxf(mx, __shfl_xor(mx, off, 32));
      float mnew  = fmaxf(mrow[v], mx);
      float alpha = __expf(mrow[v] - mnew);
      mrow[v] = mnew;
      float p0 = __expf(e0 - mnew);
      float p1 = __expf(e1 - mnew);
      float ps = p0 + p1;
#pragma unroll
      for (int off = 1; off < 16; off <<= 1)
        ps += __shfl_xor(ps, off, 32);
      lrow[v] = lrow[v] * alpha + ps;
#pragma unroll
      for (int g = 0; g < 4; ++g) o[g][v] *= alpha;
      // spill P (C-frag layout) to LDS in row-major bf16
      int prow = v + 8 * half;
      pw[prow * 40 + r]      = f2bf(p0);
      pw[prow * 40 + 16 + r] = f2bf(p1);
    }
    __syncthreads();
    // --- O += P(16x32) . V(32x64); vT rows give B fragments directly ---
    v16bf pf = frag_ld(&pw[r * 40], half);
#pragma unroll
    for (int g = 0; g < 4; ++g) {
      v16bf vb = frag_ld(vp + (size_t)(g * 16 + r) * SEQ + j0, half);
      o[g] = wmma_bf16(pf, vb, o[g]);
    }
    __syncthreads();
  }

  // epilogue: normalize and store bf16 into [B, S, H*Hd] for the out-projection
  const int b = bh >> 4, h = bh & 15;
#pragma unroll
  for (int g = 0; g < 4; ++g)
#pragma unroll
    for (int v = 0; v < 8; ++v) {
      int i = q0 + wave * 16 + v + 8 * half;
      float val = o[g][v] / lrow[v];
      attnOut[((size_t)(b * SEQ + i)) * DMODEL + h * HDIM + g * 16 + r] =
          f2bf(val);
    }
}

// ---------------------------------------------------------------------------
// Host-side orchestration
// ---------------------------------------------------------------------------
extern "C" void kernel_launch(void* const* d_in, const int* in_sizes, int n_in,
                              void* d_out, int out_size, void* d_ws,
                              size_t ws_size, hipStream_t stream) {
  const float* x    = (const float*)d_in[0];
  const float* wqkv = (const float*)d_in[1];
  const float* wo   = (const float*)d_in[2];
  float* out = (float*)d_out;

  char* ws = (char*)d_ws;
  unsigned short* xb    = (unsigned short*)(ws);                          //  8 MiB
  unsigned short* wqkvT = (unsigned short*)(ws + (size_t)(8u)  * 1048576);//  6 MiB
  unsigned short* woT   = (unsigned short*)(ws + (size_t)(14u) * 1048576);//  2 MiB
  unsigned short* qB    = (unsigned short*)(ws + (size_t)(16u) * 1048576);//  8 MiB
  unsigned short* kB    = (unsigned short*)(ws + (size_t)(24u) * 1048576);//  8 MiB
  unsigned short* vTB   = (unsigned short*)(ws + (size_t)(32u) * 1048576);//  8 MiB
  unsigned short* attnB = (unsigned short*)(ws + (size_t)(40u) * 1048576);//  8 MiB

  const int prepTotal = ROWS * DMODEL + N_QKV * DMODEL + DMODEL * DMODEL;
  prep_kernel<<<(prepTotal + 255) / 256, 256, 0, stream>>>(x, wqkv, wo, xb,
                                                           wqkvT, woT);
  gemm_kernel<0><<<dim3(N_QKV / 128, ROWS / 128), 256, 0, stream>>>(
      xb, wqkvT, N_QKV, qB, kB, vTB, nullptr);
  rope_kernel<<<(BATCH * HEADS * SEQ * 32) / 256, 256, 0, stream>>>(qB, kB);
  flash_kernel<<<dim3(SEQ / 64, BATCH * HEADS), 128, 0, stream>>>(qB, kB, vTB,
                                                                  attnB);
  gemm_kernel<1><<<dim3(DMODEL / 128, ROWS / 128), 256, 0, stream>>>(
      attnB, woT, DMODEL, nullptr, nullptr, nullptr, out);
}
